// simple_GAT_36532991820530
// MI455X (gfx1250) — matrix operand order repaired and checked by reading, not compiled, not run on hardware
//
#include <hip/hip_runtime.h>
#include <hip/hip_bf16.h>

#define N_NODES 100000
#define N_EDGES 1600000
#define E_TOT   (N_EDGES + N_NODES)
#define DH      128
#define N_LAYERS 5
#define N_GRAPHS 64
#define NM      (N_NODES * DH)
#define GEMM_BLKS ((N_NODES + 127) / 128)        /* 782 */
#define N_PAD   (GEMM_BLKS * 128)                /* 100096 padded rows */

typedef __attribute__((ext_vector_type(16))) _Float16 v16h;
typedef __attribute__((ext_vector_type(8)))  _Float16 v8h;
typedef __attribute__((ext_vector_type(8)))  float    v8f;

union AFrag { v16h v; v8h h[2]; };

__device__ __forceinline__ int fenc(float f) {
  int i = __float_as_int(f);
  return i >= 0 ? i : (i ^ 0x7fffffff);
}
__device__ __forceinline__ float fdec(int i) {
  return __int_as_float(i >= 0 ? i : (i ^ 0x7fffffff));
}
__device__ __forceinline__ float wave_sum(float v) {
  #pragma unroll
  for (int m = 16; m > 0; m >>= 1) v += __shfl_xor(v, m, 32);
  return v;
}
// relaxed agent-scope f32 add -> native global_atomic_add_f32
__device__ __forceinline__ void fadd_atomic(float* p, float v) {
  __hip_atomic_fetch_add(p, v, __ATOMIC_RELAXED, __HIP_MEMORY_SCOPE_AGENT);
}

// ---- convert h (f32) -> h16 (f16), row-major [N,128] ----
__global__ __launch_bounds__(256) void k_h16(const float* __restrict__ hin,
                                             _Float16* __restrict__ h16, int n) {
  int tid = blockIdx.x * 256 + threadIdx.x;
  if (tid < n) h16[tid] = (_Float16)hin[tid];
}

// ---- convert W[l] (f32, K x N row-major) -> Wt16 (f16, N x K row-major) ----
__global__ __launch_bounds__(256) void k_wt(const float* __restrict__ Wl,
                                            _Float16* __restrict__ wt) {
  int tid = blockIdx.x * 256 + threadIdx.x;   // tid = n*128 + k
  int n = tid >> 7, k = tid & 127;
  wt[tid] = (_Float16)Wl[k * DH + n];
}

// ---- xp = h @ W via v_wmma_f32_16x16x32_f16 ----
// block: 256 threads = 8 waves; wave w computes rows [blk*128 + w*16, +16) x 128 cols.
// Wt16 (32KB) is staged into LDS once per block via async global->LDS copies.
__global__ __launch_bounds__(256) void k_gemm(const _Float16* __restrict__ h16,
                                              const _Float16* __restrict__ wt16,
                                              float* __restrict__ xp) {
  __shared__ __align__(32) _Float16 sW[DH * DH];   // 32 KB, [N,K] row-major

  // async copy Wt16 -> LDS: 2048 x 16B chunks, 8 per thread
  {
    unsigned ldsBase = (unsigned)(size_t)&sW[0];
    #pragma unroll
    for (int it = 0; it < 8; ++it) {
      int i = it * 256 + threadIdx.x;               // chunk index
      unsigned loff = ldsBase + i * 16;
      unsigned long long ga = (unsigned long long)(size_t)(wt16 + i * 8);
      asm volatile("global_load_async_to_lds_b128 %0, %1, off"
                   :: "v"(loff), "v"(ga) : "memory");
    }
    asm volatile("s_wait_asynccnt 0x0" ::: "memory");
  }
  __syncthreads();

  const int wid  = threadIdx.x >> 5;
  const int lane = threadIdx.x & 31;
  const int hiw  = lane >> 4;          // half-wave: 0 or 1
  const long rowBase = (long)blockIdx.x * 128 + wid * 16;   // padded: always valid

  // A fragments for the 4 k-steps (K = ks*32 .. +31), kept in VGPRs across all tiles
  const _Float16* aptr = h16 + (rowBase + (lane & 15)) * DH + hiw * 8;
  AFrag a[4];
  #pragma unroll
  for (int ks = 0; ks < 4; ++ks) {
    a[ks].h[0] = *(const v8h*)(aptr + ks * 32);        // K = ks*32 + hiw*8 + [0..7]
    a[ks].h[1] = *(const v8h*)(aptr + ks * 32 + 16);   // K = ks*32 + 16 + hiw*8 + [0..7]
  }

  const int n0 = lane & 15;
  const int kh16 = hiw * 8;
  #pragma unroll 2
  for (int ct = 0; ct < 8; ++ct) {
    const _Float16* bp = sW + (ct * 16 + n0) * DH + kh16;  // LDS, 16B-aligned
    v8f c = {};
    #pragma unroll
    for (int ks = 0; ks < 4; ++ks) {
      AFrag bf; bf.h[0] = *(const v8h*)(bp + ks * 32); bf.h[1] = *(const v8h*)(bp + ks * 32 + 16);
      v16h b = bf.v;                                   // K = ks*32 + hiw*8 + [0..7], ks*32 + 16 + hiw*8 + [0..7]
      c = __builtin_amdgcn_wmma_f32_16x16x32_f16(false, a[ks].v, false, b,
                                                 (short)0, c, false, false);
    }
    float* op = xp + (rowBase + hiw * 8) * DH + ct * 16 + n0;
    #pragma unroll
    for (int v = 0; v < 8; ++v) op[v * DH] = c[v];     // row = rowBase + hiw*8 + v
  }
}

// ---- per-node attention dots: s[i] = xp[i].a_s, d[i] = xp[i].a_d (wave/node) ----
__global__ __launch_bounds__(256) void k_nodedot(const float* __restrict__ xp,
                                                 const float* __restrict__ as,
                                                 const float* __restrict__ ad,
                                                 float* __restrict__ sn,
                                                 float* __restrict__ dn) {
  int node = blockIdx.x * 8 + (threadIdx.x >> 5);
  if (node >= N_NODES) return;
  int c = (threadIdx.x & 31) * 4;
  float4 xv = *(const float4*)(xp + (long)node * DH + c);
  float4 a4 = *(const float4*)(as + c);
  float4 d4 = *(const float4*)(ad + c);
  float sp = xv.x * a4.x + xv.y * a4.y + xv.z * a4.z + xv.w * a4.w;
  float dp = xv.x * d4.x + xv.y * d4.y + xv.z * d4.z + xv.w * d4.w;
  sp = wave_sum(sp);
  dp = wave_sum(dp);
  if ((threadIdx.x & 31) == 0) { sn[node] = sp; dn[node] = dp; }
}

// ---- init: out_acc = bias (broadcast), m_enc = -inf, denom = 0 ----
__global__ __launch_bounds__(256) void k_init(float* __restrict__ outacc,
                                              const float* __restrict__ bias,
                                              int* __restrict__ menc,
                                              float* __restrict__ denom) {
  int tid = blockIdx.x * 256 + threadIdx.x;
  if (tid < NM) outacc[tid] = bias[tid & 127];
  if (tid < N_NODES) { menc[tid] = (int)0x80000000; denom[tid] = 0.0f; }
}

// ---- edge pass 1: logits + segment max ----
__global__ __launch_bounds__(256) void k_edge1(const int* __restrict__ ei,
                                               const float* __restrict__ sn,
                                               const float* __restrict__ dn,
                                               float* __restrict__ ev,
                                               int* __restrict__ menc) {
  int tid = blockIdx.x * 256 + threadIdx.x;
  if (tid >= E_TOT) return;
  int src, dst;
  if (tid < N_EDGES) { src = ei[tid]; dst = ei[N_EDGES + tid]; }
  else               { src = dst = tid - N_EDGES; }
  float e = sn[src] + dn[dst];
  e = e > 0.0f ? e : 0.2f * e;                // leaky_relu(0.2)
  ev[tid] = e;
  atomicMax(menc + dst, fenc(e));
}

// ---- edge pass 2: w = exp(e - max), denom += w ----
__global__ __launch_bounds__(256) void k_edge2(const int* __restrict__ ei,
                                               const float* __restrict__ ev,
                                               const int* __restrict__ menc,
                                               float* __restrict__ wv,
                                               float* __restrict__ denom) {
  int tid = blockIdx.x * 256 + threadIdx.x;
  if (tid >= E_TOT) return;
  int dst = (tid < N_EDGES) ? ei[N_EDGES + tid] : tid - N_EDGES;
  float w = __expf(ev[tid] - fdec(menc[dst]));
  wv[tid] = w;
  fadd_atomic(denom + dst, w);
}

// ---- edge pass 3: out[dst] += alpha * xp[src]  (wave/edge, 4 cols/lane) ----
__global__ __launch_bounds__(256) void k_edge3(const int* __restrict__ ei,
                                               const float* __restrict__ wv,
                                               const float* __restrict__ denom,
                                               const float* __restrict__ xp,
                                               float* __restrict__ outacc) {
  int eidx = blockIdx.x * 8 + (threadIdx.x >> 5);
  if (eidx >= E_TOT) return;
  int src, dst;
  if (eidx < N_EDGES) { src = ei[eidx]; dst = ei[N_EDGES + eidx]; }
  else                { src = dst = eidx - N_EDGES; }
  float alpha = wv[eidx] / denom[dst];
  int c = (threadIdx.x & 31) * 4;
  float4 xv = *(const float4*)(xp + (long)src * DH + c);
  float* ob = outacc + (long)dst * DH + c;
  fadd_atomic(ob + 0, alpha * xv.x);
  fadd_atomic(ob + 1, alpha * xv.y);
  fadd_atomic(ob + 2, alpha * xv.z);
  fadd_atomic(ob + 3, alpha * xv.w);
}

// ---- relu + batchnorm (eval mode), in place ----
__global__ __launch_bounds__(256) void k_post(float* __restrict__ outacc,
                                              const float* __restrict__ gam,
                                              const float* __restrict__ bet,
                                              const float* __restrict__ mu,
                                              const float* __restrict__ var) {
  int tid = blockIdx.x * 256 + threadIdx.x;
  if (tid >= NM) return;
  int c = tid & 127;
  float v = fmaxf(outacc[tid], 0.0f);
  outacc[tid] = (v - mu[c]) * (gam[c] * rsqrtf(var[c] + 1e-5f)) + bet[c];
}

// ---- zero pool accumulators ----
__global__ __launch_bounds__(256) void k_zero(float* __restrict__ gsum,
                                              float* __restrict__ gcnt) {
  int tid = blockIdx.x * 256 + threadIdx.x;
  if (tid < N_GRAPHS * DH) gsum[tid] = 0.0f;
  if (tid < N_GRAPHS) gcnt[tid] = 0.0f;
}

// ---- global mean pool: accumulate (wave/node) ----
__global__ __launch_bounds__(256) void k_pool(const float* __restrict__ h,
                                              const int* __restrict__ batch,
                                              float* __restrict__ gsum,
                                              float* __restrict__ gcnt) {
  int node = blockIdx.x * 8 + (threadIdx.x >> 5);
  if (node >= N_NODES) return;
  int g = batch[node];
  int c = (threadIdx.x & 31) * 4;
  float4 xv = *(const float4*)(h + (long)node * DH + c);
  float* gb = gsum + g * DH + c;
  fadd_atomic(gb + 0, xv.x);
  fadd_atomic(gb + 1, xv.y);
  fadd_atomic(gb + 2, xv.z);
  fadd_atomic(gb + 3, xv.w);
  if ((threadIdx.x & 31) == 0) fadd_atomic(gcnt + g, 1.0f);
}

// ---- final linear: out[g] = (gsum[g]/cnt[g]) . lin_w + lin_b (wave/graph) ----
__global__ __launch_bounds__(256) void k_final(const float* __restrict__ gsum,
                                               const float* __restrict__ gcnt,
                                               const float* __restrict__ lw,
                                               const float* __restrict__ lb,
                                               float* __restrict__ out) {
  int g = blockIdx.x * 8 + (threadIdx.x >> 5);
  if (g >= N_GRAPHS) return;
  int c = (threadIdx.x & 31) * 4;
  float inv = 1.0f / fmaxf(gcnt[g], 1.0f);
  float4 sv = *(const float4*)(gsum + g * DH + c);
  float4 w4 = *(const float4*)(lw + c);
  float p = (sv.x * w4.x + sv.y * w4.y + sv.z * w4.z + sv.w * w4.w) * inv;
  p = wave_sum(p);
  if ((threadIdx.x & 31) == 0) out[g] = p + lb[0];
}

extern "C" void kernel_launch(void* const* d_in, const int* in_sizes, int n_in,
                              void* d_out, int out_size, void* d_ws, size_t ws_size,
                              hipStream_t stream) {
  const float* x     = (const float*)d_in[0];
  const int*   ei    = (const int*)  d_in[1];
  const int*   batch = (const int*)  d_in[2];
  const float* W     = (const float*)d_in[3];
  const float* a_src = (const float*)d_in[4];
  const float* a_dst = (const float*)d_in[5];
  const float* bvec  = (const float*)d_in[6];
  const float* gam   = (const float*)d_in[7];
  const float* bet   = (const float*)d_in[8];
  const float* mu    = (const float*)d_in[9];
  const float* var   = (const float*)d_in[10];
  const float* lw    = (const float*)d_in[11];
  const float* lb    = (const float*)d_in[12];
  float* out = (float*)d_out;

  char* p = (char*)d_ws;
  auto alloc = [&](size_t bytes) -> char* {
    char* r = p; p += (bytes + 255) & ~(size_t)255; return r;
  };
  float*    bufA  = (float*)   alloc((size_t)NM * 4);
  float*    bufB  = (float*)   alloc((size_t)NM * 4);
  float*    xp    = (float*)   alloc((size_t)N_PAD * DH * 4);   // padded rows
  _Float16* h16   = (_Float16*)alloc((size_t)N_PAD * DH * 2);   // padded rows
  _Float16* wt16  = (_Float16*)alloc((size_t)DH * DH * 2);
  float*    sn    = (float*)   alloc((size_t)N_NODES * 4);
  float*    dn    = (float*)   alloc((size_t)N_NODES * 4);
  int*      menc  = (int*)     alloc((size_t)N_NODES * 4);
  float*    denom = (float*)   alloc((size_t)N_NODES * 4);
  float*    ev    = (float*)   alloc((size_t)E_TOT * 4);
  float*    wv    = (float*)   alloc((size_t)E_TOT * 4);
  float*    gsum  = (float*)   alloc((size_t)N_GRAPHS * DH * 4);
  float*    gcnt  = (float*)   alloc((size_t)N_GRAPHS * 4);

  const int gNM    = (NM + 255) / 256;             // 50000
  const int gNodeW = (N_NODES + 7) / 8;            // 12500 (wave/node)
  const int gEdge  = (E_TOT + 255) / 256;          // 6641
  const int gEdgeW = (E_TOT + 7) / 8;              // 212500 (wave/edge)

  const float* hin = x;
  for (int l = 0; l < N_LAYERS; ++l) {
    float* outacc = (l & 1) ? bufB : bufA;
    k_h16 <<<gNM,       256, 0, stream>>>(hin, h16, NM);
    k_wt  <<<64,        256, 0, stream>>>(W + (size_t)l * DH * DH, wt16);
    k_gemm<<<GEMM_BLKS, 256, 0, stream>>>(h16, wt16, xp);
    k_nodedot<<<gNodeW, 256, 0, stream>>>(xp, a_src + l * DH, a_dst + l * DH, sn, dn);
    k_init <<<gNM,   256, 0, stream>>>(outacc, bvec + l * DH, menc, denom);
    k_edge1<<<gEdge, 256, 0, stream>>>(ei, sn, dn, ev, menc);
    k_edge2<<<gEdge, 256, 0, stream>>>(ei, ev, menc, wv, denom);
    k_edge3<<<gEdgeW,256, 0, stream>>>(ei, wv, denom, xp, outacc);
    k_post <<<gNM,   256, 0, stream>>>(outacc, gam + l * DH, bet + l * DH,
                                       mu + l * DH, var + l * DH);
    hin = outacc;
  }
  k_zero <<<32,     256, 0, stream>>>(gsum, gcnt);
  k_pool <<<gNodeW, 256, 0, stream>>>(hin, batch, gsum, gcnt);
  k_final<<<8,      256, 0, stream>>>(gsum, gcnt, lw, lb, out);
}
